// RWKV6Block_42743514530298
// MI455X (gfx1250) — matrix operand order repaired and checked by reading, not compiled, not run on hardware
//
#include <hip/hip_runtime.h>
#include <hip/hip_bf16.h>

// ---------------- problem constants (from reference) ----------------
#define BB 4
#define TT 2048
#define DD 1024
#define HH 16
#define DK 64
#define LR 64
#define FF 4096
#define MM (BB*TT)          // 8192 rows
#define CL 8                // scan chunk length
#define NCH (TT/CL)         // 256 chunks per sequence
#define BH (BB*HH)          // 64 (b,h) pairs

typedef __bf16 v16bf __attribute__((ext_vector_type(16)));
typedef float  v8f   __attribute__((ext_vector_type(8)));
typedef unsigned short u16;

__device__ __forceinline__ u16 f2bf(float f) {
    unsigned int u = __float_as_uint(f);
    u += 0x7FFFu + ((u >> 16) & 1u);      // round-to-nearest-even
    return (u16)(u >> 16);
}

// CDNA5 async global->LDS copy (ASYNCcnt-tracked), GV addressing mode
__device__ __forceinline__ void async_b128(unsigned lds_off, const void* gptr) {
    asm volatile("global_load_async_to_lds_b128 %0, %1, off"
                 :: "v"(lds_off), "v"(gptr) : "memory");
}
__device__ __forceinline__ void wait_async0() {
    asm volatile("s_wait_asynccnt 0x0" ::: "memory");
}
__device__ __forceinline__ void wait_async4() {
    asm volatile("s_wait_asynccnt 0x4" ::: "memory");
}

// ---------------- weight convert + transpose: W[R,C] f32 -> Wt[C,R] bf16 ----
__global__ void k_transpose_bf16(const float* __restrict__ W, u16* __restrict__ Wt,
                                 int R, int C) {
    __shared__ float tile[32][33];
    int c0 = blockIdx.x * 32, r0 = blockIdx.y * 32;
    int tx = threadIdx.x, ty = threadIdx.y;          // block (32,8)
    #pragma unroll
    for (int i = 0; i < 32; i += 8)
        tile[ty + i][tx] = W[(size_t)(r0 + ty + i) * C + (c0 + tx)];
    __syncthreads();
    #pragma unroll
    for (int i = 0; i < 32; i += 8)
        Wt[(size_t)(c0 + ty + i) * R + (r0 + tx)] = f2bf(tile[tx][ty + i]);
}

// ---------------- block reduction helper (256 threads, 2 values) ------------
__device__ __forceinline__ void block_reduce2_256(float& a, float& b) {
    __shared__ float s1[256], s2[256];
    int tid = threadIdx.x;
    s1[tid] = a; s2[tid] = b;
    __syncthreads();
    #pragma unroll
    for (int st = 128; st > 0; st >>= 1) {
        if (tid < st) { s1[tid] += s1[tid + st]; s2[tid] += s2[tid + st]; }
        __syncthreads();
    }
    a = s1[0]; b = s2[0];
    __syncthreads();
}

// ---------------- fused rmsnorm(ln1) + token-shift + 5-way mix --------------
__global__ __launch_bounds__(256)
void k_mix(const float* __restrict__ x, const float* __restrict__ ln1w,
           const float* __restrict__ mur, const float* __restrict__ muk,
           const float* __restrict__ muv, const float* __restrict__ muw,
           const float* __restrict__ mug,
           u16* __restrict__ xr, u16* __restrict__ xk, u16* __restrict__ xv,
           u16* __restrict__ xg, float* __restrict__ xw) {
    int bt = blockIdx.x, tid = threadIdx.x;
    bool prevOK = (bt & (TT - 1)) != 0;
    float4 c = ((const float4*)(x + (size_t)bt * DD))[tid];
    float4 p = make_float4(0.f, 0.f, 0.f, 0.f);
    if (prevOK) p = ((const float4*)(x + (size_t)(bt - 1) * DD))[tid];
    float ssc = c.x*c.x + c.y*c.y + c.z*c.z + c.w*c.w;
    float ssp = p.x*p.x + p.y*p.y + p.z*p.z + p.w*p.w;
    block_reduce2_256(ssc, ssp);
    float r1 = rsqrtf(ssc / DD + 1e-6f);
    float r0 = rsqrtf(ssp / DD + 1e-6f);
    float cv[4] = {c.x, c.y, c.z, c.w};
    float pv[4] = {p.x, p.y, p.z, p.w};
    #pragma unroll
    for (int e = 0; e < 4; e++) {
        int d = tid * 4 + e;
        size_t o = (size_t)bt * DD + d;
        float w  = ln1w[d];
        float h  = w * cv[e] * r1;
        float hp = prevOK ? (w * pv[e] * r0) : 0.f;
        float xx = hp - h;
        xr[o] = f2bf(h + xx * mur[d]);
        xk[o] = f2bf(h + xx * muk[d]);
        xv[o] = f2bf(h + xx * muv[d]);
        xg[o] = f2bf(h + xx * mug[d]);
        xw[o] = h + xx * muw[d];
    }
}

// ---------------- rmsnorm(ln2): f32 in -> bf16 out --------------------------
__global__ __launch_bounds__(256)
void k_rms_bf16(const float* __restrict__ xin, const float* __restrict__ w,
                u16* __restrict__ out) {
    int bt = blockIdx.x, tid = threadIdx.x;
    float4 c = ((const float4*)(xin + (size_t)bt * DD))[tid];
    float ss = c.x*c.x + c.y*c.y + c.z*c.z + c.w*c.w, dummy = 0.f;
    block_reduce2_256(ss, dummy);
    float r1 = rsqrtf(ss / DD + 1e-6f);
    float cv[4] = {c.x, c.y, c.z, c.w};
    #pragma unroll
    for (int e = 0; e < 4; e++) {
        int d = tid * 4 + e;
        out[(size_t)bt * DD + d] = f2bf(w[d] * cv[e] * r1);
    }
}

// ---------------- bf16 WMMA GEMM: C[M,N] = A[M,K] @ Bt[N,K]^T ---------------
// Async-to-LDS double-buffered pipeline.  ACT: 0 none, 1 silu, 2 gelu(erf).
template<int ACT, bool RES, bool OBF16>
__global__ __launch_bounds__(256)
void k_gemm_bf16(const u16* __restrict__ A, const u16* __restrict__ Bt,
                 float* __restrict__ Cf, u16* __restrict__ Cb,
                 const float* __restrict__ Res, int M_, int N_, int K_) {
    constexpr int LDT = 40;                 // padded LDS row stride (elems)
    __shared__ u16 sA[2][128 * LDT];
    __shared__ u16 sB[2][128 * LDT];
    union Fr { uint4 q[2]; v16bf v; };

    int tid  = threadIdx.x;
    int wave = tid >> 5, lane = tid & 31;
    int wm = wave & 3, wn = wave >> 2;      // 4 x 2 wave grid
    int bm = blockIdx.y * 128, bn = blockIdx.x * 128;
    int lrow = lane & 15, lhalf = lane >> 4;

    int lrowA = tid >> 2, lcol = (tid & 3) * 8;   // staging: 2 rows per 8 thr grp
    auto issue_tile = [&](int kc, int buf) {
        #pragma unroll
        for (int i = 0; i < 2; i++) {
            int row = lrowA + i * 64;
            async_b128((unsigned)(size_t)&sA[buf][row * LDT + lcol],
                       A  + (size_t)(bm + row) * K_ + kc + lcol);
            async_b128((unsigned)(size_t)&sB[buf][row * LDT + lcol],
                       Bt + (size_t)(bn + row) * K_ + kc + lcol);
        }
    };

    v8f acc[2][4];
    #pragma unroll
    for (int a = 0; a < 2; a++)
        #pragma unroll
        for (int b = 0; b < 4; b++)
            #pragma unroll
            for (int j = 0; j < 8; j++) acc[a][b][j] = 0.f;

    int nk = K_ >> 5;
    issue_tile(0, 0);
    for (int kt = 0; kt < nk; kt++) {
        int buf = kt & 1;
        if (kt + 1 < nk) { issue_tile((kt + 1) << 5, buf ^ 1); wait_async4(); }
        else             { wait_async0(); }
        __syncthreads();                     // publish current tile

        v16bf afr[2];
        #pragma unroll
        for (int tm = 0; tm < 2; tm++) {
            int row = wm * 32 + tm * 16 + lrow;
            const uint4* p = (const uint4*)(&sA[buf][row * LDT + lhalf * 8]);
            Fr f; f.q[0] = p[0]; f.q[1] = p[2];      // +32B = K+16
            afr[tm] = f.v;
        }
        v16bf bfr[4];
        #pragma unroll
        for (int tn = 0; tn < 4; tn++) {
            int row = wn * 64 + tn * 16 + lrow;
            const uint4* p = (const uint4*)(&sB[buf][row * LDT + lhalf * 8]);
            Fr f; f.q[0] = p[0]; f.q[1] = p[2];
            bfr[tn] = f.v;
        }
        #pragma unroll
        for (int tm = 0; tm < 2; tm++)
            #pragma unroll
            for (int tn = 0; tn < 4; tn++)
                acc[tm][tn] = __builtin_amdgcn_wmma_f32_16x16x32_bf16(
                    false, afr[tm], false, bfr[tn], (short)0, acc[tm][tn],
                    false, false);
        __syncthreads();                     // all waves done reading buf
    }

    #pragma unroll
    for (int tm = 0; tm < 2; tm++)
        #pragma unroll
        for (int tn = 0; tn < 4; tn++)
            #pragma unroll
            for (int j = 0; j < 8; j++) {
                int m = bm + wm * 32 + tm * 16 + lhalf * 8 + j;
                int n = bn + wn * 64 + tn * 16 + lrow;
                size_t o = (size_t)m * N_ + n;
                float v = acc[tm][tn][j];
                if (ACT == 1) v = v / (1.f + __expf(-v));
                else if (ACT == 2) v = 0.5f * v * (1.f + erff(v * 0.70710678f));
                if (RES) v += Res[o];
                if (OBF16) Cb[o] = f2bf(v);
                else       Cf[o] = v;
            }
}

// ---------------- LoRA: a1 = tanh(xw @ wA), split-K --------------------------
__global__ __launch_bounds__(256)
void k_lora1(const float* __restrict__ xw, const float* __restrict__ wA,
             float* __restrict__ a1) {
    __shared__ float red[256];
    int m = blockIdx.x, n = threadIdx.x & 63, q = threadIdx.x >> 6;
    const float* ar = xw + (size_t)m * DD + q * 256;
    float acc = 0.f;
    #pragma unroll 4
    for (int k2 = 0; k2 < 256; k2++) acc += ar[k2] * wA[(q * 256 + k2) * LR + n];
    red[threadIdx.x] = acc;
    __syncthreads();
    if (q == 0)
        a1[(size_t)m * LR + n] = tanhf(red[n] + red[64 + n] + red[128 + n] + red[192 + n]);
}

// ---------------- LoRA2 + decay: decay = exp(-exp(w0 + a1@wB)) --------------
// decay floored at 6e-5 so chunk-of-8 cumulative products stay in f32 range.
__global__ __launch_bounds__(256)
void k_lora2(const float* __restrict__ a1, const float* __restrict__ wB,
             const float* __restrict__ w0, float* __restrict__ dcy) {
    int m = blockIdx.x, n = blockIdx.y * 256 + threadIdx.x;
    const float* ar = a1 + (size_t)m * LR;
    float acc = 0.f;
    #pragma unroll
    for (int k2 = 0; k2 < LR; k2++) acc += ar[k2] * wB[k2 * DD + n];
    float w = w0[n] + acc;
    dcy[(size_t)m * DD + n] = fmaxf(expf(-expf(w)), 6e-5f);
}

// ---------------- chunked scan, phase 1 (fully parallel) --------------------
// Per chunk (b,h,c): cumprods a_t, r~ = r*a_{t-1}, k~ = k/a_t, k^ = k*aL/a_t.
// o_intra = (masked r~ k~^T + diag(r.(u*k))) V ;  U = k^T V ; emit aL, r~.
__global__ __launch_bounds__(64)
void k_chunk(const float* __restrict__ rb, const float* __restrict__ kb,
             const float* __restrict__ vb, const float* __restrict__ dcy,
             const float* __restrict__ uw, float* __restrict__ o,
             float* __restrict__ rtbuf, float* __restrict__ Ubuf,
             float* __restrict__ albuf) {
    int blk = blockIdx.x;                 // bh*NCH + c
    int c = blk % NCH, bh = blk / NCH;
    int b = bh >> 4, h = bh & 15;
    int j = threadIdx.x;
    __shared__ float Rt[CL][64], Kt[CL][64], RUK[CL][64], KhT[64][CL], Pm[CL][CL];

    size_t base = ((size_t)(b * TT + c * CL)) * DD + h * 64 + j;
    float uu = uw[h * 64 + j];
    float rv[CL], kv[CL], vv[CL], A[CL];
    float a = 1.f;
    #pragma unroll
    for (int t = 0; t < CL; t++) {
        size_t off = base + (size_t)t * DD;
        rv[t] = rb[off]; kv[t] = kb[off]; vv[t] = vb[off];
        a *= dcy[off];
        A[t] = a;
    }
    float aL = A[CL - 1];
    #pragma unroll
    for (int t = 0; t < CL; t++) {
        float ap = t ? A[t - 1] : 1.f;
        float rt = rv[t] * ap;
        Rt[t][j] = rt;
        rtbuf[base + (size_t)t * DD] = rt;
        Kt[t][j]  = kv[t] / A[t];
        KhT[j][t] = kv[t] * (aL / A[t]);
        RUK[t][j] = rv[t] * uu * kv[t];
    }
    __syncthreads();
    {   // P[t][s]: strict-lower = r~_t . k~_s ; diagonal = r_t.(u*k_t); upper=0
        int t = j >> 3, s = j & 7;
        float pv = 0.f;
        if (s < t) {
            #pragma unroll
            for (int i = 0; i < 64; i += 4) {
                float4 rr = *(const float4*)&Rt[t][i];
                float4 kk = *(const float4*)&Kt[s][i];
                pv += rr.x*kk.x + rr.y*kk.y + rr.z*kk.z + rr.w*kk.w;
            }
        } else if (s == t) {
            #pragma unroll
            for (int i = 0; i < 64; i += 4) {
                float4 rr = *(const float4*)&RUK[t][i];
                pv += rr.x + rr.y + rr.z + rr.w;
            }
        }
        Pm[t][s] = pv;
    }
    __syncthreads();
    #pragma unroll
    for (int t = 0; t < CL; t++) {          // o_intra
        float acc = 0.f;
        #pragma unroll
        for (int s = 0; s <= t; s++) acc += Pm[t][s] * vv[s];
        o[base + (size_t)t * DD] = acc;
    }
    // U[:, j] = sum_t k^_t v_t[j]
    float* Up = Ubuf + (size_t)blk * 4096 + j * 64;
    #pragma unroll
    for (int i0 = 0; i0 < 64; i0 += 4) {
        float r0 = 0.f, r1 = 0.f, r2 = 0.f, r3 = 0.f;
        #pragma unroll
        for (int t = 0; t < CL; t++) {
            float vt = vv[t];
            r0 += KhT[i0    ][t] * vt;
            r1 += KhT[i0 + 1][t] * vt;
            r2 += KhT[i0 + 2][t] * vt;
            r3 += KhT[i0 + 3][t] * vt;
        }
        *(float4*)(Up + i0) = make_float4(r0, r1, r2, r3);
    }
    albuf[(size_t)blk * 64 + j] = aL;
}

// ---------------- chunked scan, phase 2 (cheap sequential carry) ------------
// S_c = diag(aL_c) S_{c-1} + U_c ; pre-update S written over U slot (aliased).
__global__ __launch_bounds__(64)
void k_carry(float* __restrict__ USbuf, const float* __restrict__ albuf) {
    int bh = blockIdx.x, j = threadIdx.x;
    __shared__ float al_s[64];
    union F64 { float4 q[16]; float f[64]; };
    float S[64];
    #pragma unroll
    for (int i = 0; i < 64; i++) S[i] = 0.f;
    F64 Ub;
    const float* Up0 = USbuf + (size_t)bh * NCH * 4096 + j * 64;
    #pragma unroll
    for (int i = 0; i < 16; i++) Ub.q[i] = ((const float4*)Up0)[i];
    float aln = albuf[(size_t)bh * NCH * 64 + j];
    for (int c = 0; c < NCH; c++) {
        F64 Uc = Ub;                         // consume prefetched U_c
        float alc = aln;
        if (c + 1 < NCH) {                   // prefetch U_{c+1}
            const float* Upn = USbuf + ((size_t)bh * NCH + c + 1) * 4096 + j * 64;
            #pragma unroll
            for (int i = 0; i < 16; i++) Ub.q[i] = ((const float4*)Upn)[i];
            aln = albuf[((size_t)bh * NCH + c + 1) * 64 + j];
        }
        // write S (state BEFORE chunk c) over the consumed U_c slot
        float* Sp = USbuf + ((size_t)bh * NCH + c) * 4096 + j * 64;
        #pragma unroll
        for (int i = 0; i < 16; i++)
            ((float4*)Sp)[i] = make_float4(S[4*i], S[4*i+1], S[4*i+2], S[4*i+3]);
        al_s[j] = alc;
        __syncthreads();
        #pragma unroll
        for (int i = 0; i < 64; i++) S[i] = al_s[i] * S[i] + Uc.f[i];
        __syncthreads();
    }
}

// ---------------- chunked scan, phase 3 (fully parallel cross term) ---------
// o[t] += r~_t @ S_{c-1}
__global__ __launch_bounds__(64)
void k_cross(const float* __restrict__ Sbuf, const float* __restrict__ rtbuf,
             float* __restrict__ o) {
    int blk = blockIdx.x;
    int c = blk % NCH, bh = blk / NCH;
    int b = bh >> 4, h = bh & 15;
    int j = threadIdx.x;
    __shared__ float Rs[CL][64];
    union F64 { float4 q[16]; float f[64]; } Sc;
    const float* Sp = Sbuf + (size_t)blk * 4096 + j * 64;
    #pragma unroll
    for (int i = 0; i < 16; i++) Sc.q[i] = ((const float4*)Sp)[i];
    size_t base = ((size_t)(b * TT + c * CL)) * DD + h * 64 + j;
    #pragma unroll
    for (int t = 0; t < CL; t++) Rs[t][j] = rtbuf[base + (size_t)t * DD];
    __syncthreads();
    #pragma unroll
    for (int t = 0; t < CL; t++) {
        float acc = 0.f;
        #pragma unroll
        for (int i = 0; i < 64; i += 4) {
            float4 rr = *(const float4*)&Rs[t][i];
            acc += rr.x*Sc.f[i] + rr.y*Sc.f[i+1] + rr.z*Sc.f[i+2] + rr.w*Sc.f[i+3];
        }
        size_t oo = base + (size_t)t * DD;
        o[oo] += acc;                        // single owner per (t,j): safe RMW
    }
}

// ---------------- per-head group norm * silu-gate -> bf16 -------------------
__global__ __launch_bounds__(64)
void k_gnorm(const float* __restrict__ o, const float* __restrict__ g,
             const float* __restrict__ gnw, const float* __restrict__ gnb,
             u16* __restrict__ obar) {
    int bth = blockIdx.x;
    int h = bth & 15, bt = bth >> 4;
    int j = threadIdx.x;
    __shared__ float s1[64], s2[64];
    size_t off = (size_t)bt * DD + h * 64;
    float xv = o[off + j];
    s1[j] = xv; s2[j] = xv * xv;
    __syncthreads();
    #pragma unroll
    for (int st = 32; st > 0; st >>= 1) {
        if (j < st) { s1[j] += s1[j + st]; s2[j] += s2[j + st]; }
        __syncthreads();
    }
    float mean = s1[0] * (1.f / 64.f);
    float var  = s2[0] * (1.f / 64.f) - mean * mean;
    float xn = (xv - mean) * rsqrtf(var + 1e-5f) * gnw[h * 64 + j] + gnb[h * 64 + j];
    obar[off + j] = f2bf(xn * g[off + j]);   // g already has silu applied
}

// ---------------- launcher ---------------------------------------------------
extern "C" void kernel_launch(void* const* d_in, const int* in_sizes, int n_in,
                              void* d_out, int out_size, void* d_ws, size_t ws_size,
                              hipStream_t stream) {
    const float* x     = (const float*)d_in[0];
    const float* ln1_w = (const float*)d_in[1];
    const float* mu_r  = (const float*)d_in[2];
    const float* mu_k  = (const float*)d_in[3];
    const float* mu_v  = (const float*)d_in[4];
    const float* mu_w  = (const float*)d_in[5];
    const float* mu_g  = (const float*)d_in[6];
    const float* Wr    = (const float*)d_in[7];
    const float* Wk    = (const float*)d_in[8];
    const float* Wv    = (const float*)d_in[9];
    const float* Wg    = (const float*)d_in[10];
    const float* Wo    = (const float*)d_in[11];
    const float* w0    = (const float*)d_in[12];
    const float* wA    = (const float*)d_in[13];
    const float* wB    = (const float*)d_in[14];
    const float* u     = (const float*)d_in[15];
    const float* gn_w  = (const float*)d_in[16];
    const float* gn_b  = (const float*)d_in[17];
    const float* ln2_w = (const float*)d_in[18];
    const float* fw1   = (const float*)d_in[19];
    const float* fw2   = (const float*)d_in[20];
    float* out = (float*)d_out;

    char* p = (char*)d_ws;
    auto alloc = [&](size_t bytes) {
        char* q = p;
        p += (bytes + 255) & ~(size_t)255;
        return (void*)q;
    };
    u16* WrT = (u16*)alloc((size_t)DD * DD * 2);
    u16* WkT = (u16*)alloc((size_t)DD * DD * 2);
    u16* WvT = (u16*)alloc((size_t)DD * DD * 2);
    u16* WgT = (u16*)alloc((size_t)DD * DD * 2);
    u16* WoT = (u16*)alloc((size_t)DD * DD * 2);
    u16* W1T = (u16*)alloc((size_t)DD * FF * 2);
    u16* W2T = (u16*)alloc((size_t)DD * FF * 2);
    u16* xr  = (u16*)alloc((size_t)MM * DD * 2);
    u16* xk  = (u16*)alloc((size_t)MM * DD * 2);
    u16* xv  = (u16*)alloc((size_t)MM * DD * 2);
    u16* xg  = (u16*)alloc((size_t)MM * DD * 2);
    float* xw  = (float*)alloc((size_t)MM * DD * 4);   // later reused as rtbuf
    float* rb  = (float*)alloc((size_t)MM * DD * 4);
    float* kb  = (float*)alloc((size_t)MM * DD * 4);
    float* vb  = (float*)alloc((size_t)MM * DD * 4);
    float* gb  = (float*)alloc((size_t)MM * DD * 4);
    float* a1  = (float*)alloc((size_t)MM * LR * 4);
    float* dcy = (float*)alloc((size_t)MM * DD * 4);
    float* ob  = (float*)alloc((size_t)MM * DD * 4);
    u16* obar = (u16*)alloc((size_t)MM * DD * 2);
    float* x1  = (float*)alloc((size_t)MM * DD * 4);
    u16* h2   = (u16*)alloc((size_t)MM * DD * 2);
    u16* t1   = (u16*)alloc((size_t)MM * FF * 2);
    float* USbuf = (float*)alloc((size_t)BH * NCH * 4096 * 4);  // U, then S
    float* albuf = (float*)alloc((size_t)BH * NCH * 64 * 4);
    float* rtbuf = xw;   // xw fully consumed by k_lora1 before k_chunk runs

    dim3 tb(32, 8);
    k_transpose_bf16<<<dim3(DD/32, DD/32), tb, 0, stream>>>(Wr, WrT, DD, DD);
    k_transpose_bf16<<<dim3(DD/32, DD/32), tb, 0, stream>>>(Wk, WkT, DD, DD);
    k_transpose_bf16<<<dim3(DD/32, DD/32), tb, 0, stream>>>(Wv, WvT, DD, DD);
    k_transpose_bf16<<<dim3(DD/32, DD/32), tb, 0, stream>>>(Wg, WgT, DD, DD);
    k_transpose_bf16<<<dim3(DD/32, DD/32), tb, 0, stream>>>(Wo, WoT, DD, DD);
    k_transpose_bf16<<<dim3(FF/32, DD/32), tb, 0, stream>>>(fw1, W1T, DD, FF);
    k_transpose_bf16<<<dim3(DD/32, FF/32), tb, 0, stream>>>(fw2, W2T, FF, DD);

    k_mix<<<MM, 256, 0, stream>>>(x, ln1_w, mu_r, mu_k, mu_v, mu_w, mu_g,
                                  xr, xk, xv, xg, xw);

    dim3 gD(DD/128, MM/128);
    k_gemm_bf16<0,false,false><<<gD, 256, 0, stream>>>(xr, WrT, rb, nullptr, nullptr, MM, DD, DD);
    k_gemm_bf16<0,false,false><<<gD, 256, 0, stream>>>(xk, WkT, kb, nullptr, nullptr, MM, DD, DD);
    k_gemm_bf16<0,false,false><<<gD, 256, 0, stream>>>(xv, WvT, vb, nullptr, nullptr, MM, DD, DD);
    k_gemm_bf16<1,false,false><<<gD, 256, 0, stream>>>(xg, WgT, gb, nullptr, nullptr, MM, DD, DD);

    k_lora1<<<MM, 256, 0, stream>>>(xw, wA, a1);
    k_lora2<<<dim3(MM, DD/256), 256, 0, stream>>>(a1, wB, w0, dcy);

    // chunked RWKV scan: parallel intra-chunk, tiny sequential carry, parallel cross
    k_chunk<<<BH*NCH, 64, 0, stream>>>(rb, kb, vb, dcy, u, ob, rtbuf, USbuf, albuf);
    k_carry<<<BH, 64, 0, stream>>>(USbuf, albuf);
    k_cross<<<BH*NCH, 64, 0, stream>>>(USbuf, rtbuf, ob);

    k_gnorm<<<MM*HH, 64, 0, stream>>>(ob, gb, gn_w, gn_b, obar);

    k_gemm_bf16<0,true,false><<<gD, 256, 0, stream>>>(obar, WoT, x1, nullptr, x, MM, DD, DD);

    k_rms_bf16<<<MM, 256, 0, stream>>>(x1, ln2_w, h2);
    k_gemm_bf16<2,false,true><<<dim3(FF/128, MM/128), 256, 0, stream>>>(h2, W1T, nullptr, t1, nullptr, MM, FF, DD);
    k_gemm_bf16<0,true,false><<<gD, 256, 0, stream>>>(t1, W2T, out, nullptr, x1, MM, DD, FF);
}
